// Decoder_58325655880211
// MI455X (gfx1250) — compile-verified, matrix operand order checked by reading, not run on hardware
//
#include <hip/hip_runtime.h>
#include <hip/hip_bf16.h>

// ---------------------------------------------------------------------------
// Show-attend-tell LSTM decoder for MI455X (gfx1250, wave32, WMMA).
// All GEMMs use v_wmma_f32_16x16x32_bf16 (fp32 accumulate), 64x16 block/wave.
// B=64, P=196, ENC=2048, ATT=512, DEC=512, EMB=512, VOCAB=10000, T=49
// ---------------------------------------------------------------------------

typedef __attribute__((ext_vector_type(16))) __bf16 v16bf;
typedef __attribute__((ext_vector_type(8)))  float  v8f;

union FragBF {
    v16bf v;
    unsigned int u[8];
};

__device__ __forceinline__ unsigned short f32_to_bf16(float f) {
    unsigned int x = __float_as_uint(f);
    unsigned int r = (x + 0x7FFFu + ((x >> 16) & 1u)) >> 16;  // RNE
    return (unsigned short)r;
}

__device__ __forceinline__ float sigf(float x) { return 1.0f / (1.0f + expf(-x)); }

// ---------------------------------------------------------------------------
// WMMA GEMM: C[M][N] = A[M][K] * Bt[N][K]^T (+bias[n]) (*rowscale[m])
// A, Bt are bf16 (stored as u16). M%64==0, N%16==0, K%32==0 required.
// One wave computes a 64x16 block (4 stacked 16x16 WMMA tiles) so the weight
// (Bt) fragment is loaded once and reused by 4 WMMAs -> weights stream at
// exactly 1.0 reads/element (L2-resident across the 49 recurrent steps).
// 8 waves per 256-thread block.
// ---------------------------------------------------------------------------
__global__ void gemm_wmma_bf16(const unsigned short* __restrict__ A,
                               const unsigned short* __restrict__ Bt,
                               float* __restrict__ C,
                               int M, int N, int K, int ldc,
                               const float* __restrict__ bias,
                               const float* __restrict__ rowscale)
{
    const int wave = threadIdx.x >> 5;
    const int lane = threadIdx.x & 31;
    const int tilesN = N >> 4;
    const int blocksM = M >> 6;                   // 64-row blocks
    const int tile = blockIdx.x * (blockDim.x >> 5) + wave;
    if (tile >= blocksM * tilesN) return;         // uniform per-wave exit

    const int tm = (tile / tilesN) << 6;
    const int tn = (tile % tilesN) << 4;

    const int hi   = lane >> 4;                   // 0 or 1 (lane half)
    const int l16  = lane & 15;
    const int koff = hi << 3;                     // +8 for upper half lanes

    const unsigned short* arow = A  + (size_t)(tm + l16) * (size_t)K;
    const unsigned short* brow = Bt + (size_t)(tn + l16) * (size_t)K;

    v8f acc[4] = {};
    for (int k0 = 0; k0 < K; k0 += 32) {
        // 16-bit A-matrix 16x32 layout: VGPR j holds K pair
        //   (j<4 ? 2j : 2j+8) + (laneHi ? 8 : 0)
        FragBF b;
#pragma unroll
        for (int j = 0; j < 8; ++j) {
            const int kk = k0 + koff + ((j < 4) ? (j << 1) : ((j << 1) + 8));
            b.u[j] = *(const unsigned int*)(brow + kk);
        }
        if (k0 + 64 <= K)                          // stream weights ahead
            __builtin_prefetch(brow + k0 + 64, 0, 1);

        FragBF a0, a1, a2, a3;
#pragma unroll
        for (int j = 0; j < 8; ++j) {
            const int kk = k0 + koff + ((j < 4) ? (j << 1) : ((j << 1) + 8));
            a0.u[j] = *(const unsigned int*)(arow + kk);
            a1.u[j] = *(const unsigned int*)(arow + (size_t)16 * K + kk);
            a2.u[j] = *(const unsigned int*)(arow + (size_t)32 * K + kk);
            a3.u[j] = *(const unsigned int*)(arow + (size_t)48 * K + kk);
        }
        acc[0] = __builtin_amdgcn_wmma_f32_16x16x32_bf16(
                     false, a0.v, false, b.v, (short)0, acc[0], false, false);
        acc[1] = __builtin_amdgcn_wmma_f32_16x16x32_bf16(
                     false, a1.v, false, b.v, (short)0, acc[1], false, false);
        acc[2] = __builtin_amdgcn_wmma_f32_16x16x32_bf16(
                     false, a2.v, false, b.v, (short)0, acc[2], false, false);
        acc[3] = __builtin_amdgcn_wmma_f32_16x16x32_bf16(
                     false, a3.v, false, b.v, (short)0, acc[3], false, false);
    }

    // C/D layout: lane(l16)=N column; VGPR r -> M = r + 8*laneHi
    const int n  = tn + l16;
    const float bv = bias ? bias[n] : 0.0f;
#pragma unroll
    for (int i = 0; i < 4; ++i) {
#pragma unroll
        for (int r = 0; r < 8; ++r) {
            const int m = tm + (i << 4) + (hi << 3) + r;
            float v = acc[i][r] + bv;
            if (rowscale) v *= rowscale[m];
            C[(size_t)m * (size_t)ldc + n] = v;
        }
    }
}

// ---------------------------------------------------------------------------
// Prep kernels
// ---------------------------------------------------------------------------
__global__ void k_f32_to_bf16(const float* __restrict__ in,
                              unsigned short* __restrict__ out, int n)
{
    int i = blockIdx.x * blockDim.x + threadIdx.x;
    if (i < n) out[i] = f32_to_bf16(in[i]);
}

// in: [K][N] fp32 -> out: [N][K] bf16
__global__ void k_transpose_bf16(const float* __restrict__ in,
                                 unsigned short* __restrict__ out, int K, int N)
{
    int i = blockIdx.x * blockDim.x + threadIdx.x;
    if (i >= K * N) return;
    int nn = i / K, kk = i % K;
    out[i] = f32_to_bf16(in[kk * N + nn]);
}

// wcomb[2048][3072] = [w_ih | w_hh] rows (already [N][K] layouts)
__global__ void k_build_wcomb(const float* __restrict__ w_ih,
                              const float* __restrict__ w_hh,
                              unsigned short* __restrict__ wcomb)
{
    int i = blockIdx.x * blockDim.x + threadIdx.x;
    if (i >= 2048 * 3072) return;
    int n = i / 3072, k = i % 3072;
    float v = (k < 2560) ? w_ih[n * 2560 + k] : w_hh[n * 512 + (k - 2560)];
    wcomb[i] = f32_to_bf16(v);
}

// bsum[n] = b_ih[n]+b_hh[n];  mask[t*64+b] = (t < caption_lengths[b]-1)
__global__ void k_bsum_mask(const float* __restrict__ b_ih,
                            const float* __restrict__ b_hh,
                            float* __restrict__ bsum,
                            const int* __restrict__ cap_len,
                            float* __restrict__ mask)
{
    int i = blockIdx.x * blockDim.x + threadIdx.x;
    if (i < 2048) bsum[i] = b_ih[i] + b_hh[i];
    if (i < 49 * 64) {
        int t = i / 64, b = i % 64;
        mask[i] = (t < cap_len[b] - 1) ? 1.0f : 0.0f;
    }
}

// avg_bf[b][e] = mean_p encoder_out[b][p][e]  (bf16), one block per b
__global__ void k_avg(const float* __restrict__ enc,
                      unsigned short* __restrict__ avg_bf)
{
    int b = blockIdx.x;
    for (int e = threadIdx.x; e < 2048; e += blockDim.x) {
        float s = 0.0f;
        for (int p = 0; p < 196; ++p)
            s += enc[((size_t)(b * 196 + p)) * 2048 + e];
        avg_bf[b * 2048 + e] = f32_to_bf16(s * (1.0f / 196.0f));
    }
}

// ---------------------------------------------------------------------------
// Attention: logits = relu(enc_proj + dec_proj) . att_w + att_b; softmax over
// P=196; alpha*mask -> d_out alphas; awe[b][e] = sum_p alpha*enc_out.
// One block per batch element. encoder_out (103MB fp32) is L2-resident.
// ---------------------------------------------------------------------------
__global__ void k_attn(const float* __restrict__ enc_proj,
                       const float* __restrict__ dec_proj,
                       const float* __restrict__ att_w,
                       const float* __restrict__ att_b,
                       const float* __restrict__ enc_out,
                       float* __restrict__ awe,
                       float* __restrict__ out_alphas,
                       const float* __restrict__ mask, int t)
{
    __shared__ float sl[196];
    __shared__ float red[256];
    const int b = blockIdx.x, tid = threadIdx.x;

    for (int p = tid; p < 196; p += 256) {
        const float* ep = enc_proj + ((size_t)(b * 196 + p)) * 512;
        const float* dp = dec_proj + b * 512;
        float s = att_b[0];
        for (int a = 0; a < 512; ++a) {
            float r = ep[a] + dp[a];
            s += (r > 0.0f ? r : 0.0f) * att_w[a];
        }
        sl[p] = s;
    }
    __syncthreads();

    float mx = -3.4e38f;
    for (int p = tid; p < 196; p += 256) mx = fmaxf(mx, sl[p]);
    red[tid] = mx; __syncthreads();
    for (int s = 128; s > 0; s >>= 1) {
        if (tid < s) red[tid] = fmaxf(red[tid], red[tid + s]);
        __syncthreads();
    }
    mx = red[0]; __syncthreads();

    float sum = 0.0f;
    for (int p = tid; p < 196; p += 256) {
        float e = expf(sl[p] - mx);
        sl[p] = e; sum += e;
    }
    red[tid] = sum; __syncthreads();
    for (int s = 128; s > 0; s >>= 1) {
        if (tid < s) red[tid] += red[tid + s];
        __syncthreads();
    }
    const float inv = 1.0f / red[0];
    __syncthreads();

    const float m = mask[t * 64 + b];
    for (int p = tid; p < 196; p += 256) {
        float a = sl[p] * inv;
        sl[p] = a;
        out_alphas[((size_t)b * 49 + t) * 196 + p] = a * m;
    }
    __syncthreads();

    for (int e = tid; e < 2048; e += 256) {
        float s = 0.0f;
        for (int p = 0; p < 196; ++p)
            s += sl[p] * enc_out[((size_t)(b * 196 + p)) * 2048 + e];
        awe[b * 2048 + e] = s;
    }
}

// xh[b][0:512]=emb(token); [512:2560]=sigmoid(gatelin)*awe; [2560:3072]=h (bf16)
__global__ void k_build_x(const float* __restrict__ emb,
                          const int* __restrict__ caps, int t,
                          const float* __restrict__ gatelin,
                          const float* __restrict__ awe,
                          const float* __restrict__ h,
                          unsigned short* __restrict__ xh)
{
    int i = blockIdx.x * blockDim.x + threadIdx.x;
    if (i >= 64 * 3072) return;
    int b = i / 3072, col = i % 3072;
    float v;
    if (col < 512) {
        int tok = caps[b * 50 + t];
        v = emb[(size_t)tok * 512 + col];
    } else if (col < 2560) {
        int e = col - 512;
        v = sigf(gatelin[b * 2048 + e]) * awe[b * 2048 + e];
    } else {
        v = h[b * 512 + (col - 2560)];
    }
    xh[i] = f32_to_bf16(v);
}

// LSTM cell: gates[b][0:512]=i, [512:1024]=f, [1024:1536]=g, [1536:2048]=o
__global__ void k_cell(const float* __restrict__ gates,
                       float* __restrict__ h, float* __restrict__ c,
                       unsigned short* __restrict__ h_bf,
                       unsigned short* __restrict__ hnew_bf,
                       const float* __restrict__ mask, int t)
{
    int i = blockIdx.x * blockDim.x + threadIdx.x;
    if (i >= 64 * 512) return;
    int b = i >> 9, j = i & 511;
    const float* g = gates + b * 2048;
    float ig = sigf(g[j]);
    float fg = sigf(g[512 + j]);
    float gg = tanhf(g[1024 + j]);
    float og = sigf(g[1536 + j]);
    float cn = fg * c[i] + ig * gg;
    float hn = og * tanhf(cn);
    hnew_bf[i] = f32_to_bf16(hn);            // preds use unmasked h_new
    float m = mask[t * 64 + b];
    float hv = (m > 0.0f) ? hn : h[i];
    float cv = (m > 0.0f) ? cn : c[i];
    h[i] = hv; c[i] = cv;
    h_bf[i] = f32_to_bf16(hv);               // next-step recurrent input
}

// ---------------------------------------------------------------------------
extern "C" void kernel_launch(void* const* d_in, const int* in_sizes, int n_in,
                              void* d_out, int out_size, void* d_ws, size_t ws_size,
                              hipStream_t stream)
{
    const float* encoder_out      = (const float*)d_in[0];
    const int*   encoded_captions = (const int*)  d_in[1];
    const int*   caption_lengths  = (const int*)  d_in[2];
    const float* enc_w   = (const float*)d_in[3];
    const float* enc_b   = (const float*)d_in[4];
    const float* dec_w   = (const float*)d_in[5];
    const float* dec_b   = (const float*)d_in[6];
    const float* att_w   = (const float*)d_in[7];
    const float* att_b   = (const float*)d_in[8];
    const float* h_w     = (const float*)d_in[9];
    const float* h_b     = (const float*)d_in[10];
    const float* c_w     = (const float*)d_in[11];
    const float* c_b     = (const float*)d_in[12];
    const float* fbeta_w = (const float*)d_in[13];
    const float* fbeta_b = (const float*)d_in[14];
    const float* w_ih    = (const float*)d_in[15];
    const float* w_hh    = (const float*)d_in[16];
    const float* b_ih    = (const float*)d_in[17];
    const float* b_hh    = (const float*)d_in[18];
    const float* fc_w    = (const float*)d_in[19];
    const float* fc_b    = (const float*)d_in[20];
    const float* emb     = (const float*)d_in[21];

    float* out_preds  = (float*)d_out;                           // [64][49][10000]
    float* out_alphas = (float*)d_out + (size_t)64 * 49 * 10000; // [64][49][196]

    // ---- workspace arena ----
    char* ws = (char*)d_ws;
    size_t off = 0;
    auto alloc = [&](size_t bytes) -> void* {
        void* p = ws + off;
        off = (off + bytes + 255) & ~(size_t)255;
        return p;
    };
    unsigned short* enc_bf   = (unsigned short*)alloc((size_t)12544 * 2048 * 2);
    unsigned short* enc_wT   = (unsigned short*)alloc((size_t)512 * 2048 * 2);
    unsigned short* dec_wT   = (unsigned short*)alloc((size_t)512 * 512 * 2);
    unsigned short* h_wT     = (unsigned short*)alloc((size_t)512 * 2048 * 2);
    unsigned short* c_wT     = (unsigned short*)alloc((size_t)512 * 2048 * 2);
    unsigned short* fbeta_wT = (unsigned short*)alloc((size_t)2048 * 512 * 2);
    unsigned short* fc_wT    = (unsigned short*)alloc((size_t)10000 * 512 * 2);
    unsigned short* wcomb    = (unsigned short*)alloc((size_t)2048 * 3072 * 2);
    float* bsum     = (float*)alloc(2048 * 4);
    float* maskbuf  = (float*)alloc(49 * 64 * 4);
    unsigned short* avg_bf  = (unsigned short*)alloc((size_t)64 * 2048 * 2);
    float* enc_proj = (float*)alloc((size_t)12544 * 512 * 4);
    float* hbuf     = (float*)alloc(64 * 512 * 4);
    float* cbuf     = (float*)alloc(64 * 512 * 4);
    unsigned short* h_bf    = (unsigned short*)alloc(64 * 512 * 2);
    unsigned short* hnew_bf = (unsigned short*)alloc(64 * 512 * 2);
    float* dec_proj = (float*)alloc(64 * 512 * 4);
    float* awe      = (float*)alloc(64 * 2048 * 4);
    float* gatelin  = (float*)alloc(64 * 2048 * 4);
    unsigned short* xh = (unsigned short*)alloc((size_t)64 * 3072 * 2);
    float* gates    = (float*)alloc(64 * 2048 * 4);
    (void)ws_size; (void)in_sizes; (void)n_in; (void)out_size;

    auto gemm = [&](const unsigned short* A, const unsigned short* Bt, float* C,
                    int M, int N, int K, int ldc,
                    const float* bias, const float* rowscale) {
        int tiles = (M >> 6) * (N >> 4);           // 64x16 blocks per wave
        int blocks = (tiles + 7) / 8;
        gemm_wmma_bf16<<<blocks, 256, 0, stream>>>(A, Bt, C, M, N, K, ldc,
                                                   bias, rowscale);
    };

    // ---- one-time prep ----
    {
        int n = 12544 * 2048;
        k_f32_to_bf16<<<(n + 255) / 256, 256, 0, stream>>>(encoder_out, enc_bf, n);
    }
    k_transpose_bf16<<<(2048 * 512 + 255) / 256, 256, 0, stream>>>(enc_w,   enc_wT,   2048, 512);
    k_transpose_bf16<<<( 512 * 512 + 255) / 256, 256, 0, stream>>>(dec_w,   dec_wT,    512, 512);
    k_transpose_bf16<<<(2048 * 512 + 255) / 256, 256, 0, stream>>>(h_w,     h_wT,     2048, 512);
    k_transpose_bf16<<<(2048 * 512 + 255) / 256, 256, 0, stream>>>(c_w,     c_wT,     2048, 512);
    k_transpose_bf16<<<( 512 * 2048 + 255) / 256, 256, 0, stream>>>(fbeta_w, fbeta_wT,  512, 2048);
    k_transpose_bf16<<<( 512 * 10000 + 255) / 256, 256, 0, stream>>>(fc_w,   fc_wT,     512, 10000);
    k_build_wcomb<<<(2048 * 3072 + 255) / 256, 256, 0, stream>>>(w_ih, w_hh, wcomb);
    k_bsum_mask<<<(49 * 64 + 255) / 256, 256, 0, stream>>>(b_ih, b_hh, bsum,
                                                           caption_lengths, maskbuf);
    k_avg<<<64, 256, 0, stream>>>(encoder_out, avg_bf);

    // ---- h0 / c0 ----
    gemm(avg_bf, h_wT, hbuf, 64, 512, 2048, 512, h_b, nullptr);
    gemm(avg_bf, c_wT, cbuf, 64, 512, 2048, 512, c_b, nullptr);
    k_f32_to_bf16<<<(64 * 512 + 255) / 256, 256, 0, stream>>>(hbuf, h_bf, 64 * 512);

    // ---- enc_proj (one-time big GEMM: 12544 x 512 x 2048) ----
    gemm(enc_bf, enc_wT, enc_proj, 12544, 512, 2048, 512, enc_b, nullptr);

    // ---- timestep loop ----
    for (int t = 0; t < 49; ++t) {
        gemm(h_bf, dec_wT, dec_proj, 64, 512, 512, 512, dec_b, nullptr);

        k_attn<<<64, 256, 0, stream>>>(enc_proj, dec_proj, att_w, att_b,
                                       encoder_out, awe, out_alphas, maskbuf, t);

        gemm(h_bf, fbeta_wT, gatelin, 64, 2048, 512, 2048, fbeta_b, nullptr);

        k_build_x<<<(64 * 3072 + 255) / 256, 256, 0, stream>>>(
            emb, encoded_captions, t, gatelin, awe, hbuf, xh);

        gemm(xh, wcomb, gates, 64, 2048, 3072, 2048, bsum, nullptr);

        k_cell<<<(64 * 512 + 255) / 256, 256, 0, stream>>>(
            gates, hbuf, cbuf, h_bf, hnew_bf, maskbuf, t);

        // preds: strided write into d_out [b][t][v], masked per row
        gemm(hnew_bf, fc_wT, out_preds + (size_t)t * 10000,
             64, 10000, 512, 49 * 10000, fc_b, maskbuf + t * 64);
    }
}
